// GPT_17317308137930
// MI455X (gfx1250) — matrix operand order repaired and checked by reading, not compiled, not run on hardware
//
#include <hip/hip_runtime.h>

// ---------------- CDNA5 WMMA plumbing ----------------
typedef __bf16 v16bf __attribute__((ext_vector_type(16)));
typedef float  v8f   __attribute__((ext_vector_type(8)));
typedef unsigned short ush8 __attribute__((ext_vector_type(8)));
typedef int    v4i   __attribute__((ext_vector_type(4)));

union Frag { ush8 u[2]; v16bf v; };

#ifndef __has_builtin
#define __has_builtin(x) 0
#endif

#if __has_builtin(__builtin_amdgcn_global_load_async_to_lds_b128)
#define HAVE_ASYNC_LDS 1
#else
#define HAVE_ASYNC_LDS 0
#endif

// 16B global -> LDS copy (async on CDNA5: tracked by ASYNCcnt)
__device__ __forceinline__ void async_copy16(const void* g, void* l) {
#if HAVE_ASYNC_LDS
  __builtin_amdgcn_global_load_async_to_lds_b128(
      (__attribute__((address_space(1))) v4i*)(g),
      (__attribute__((address_space(3))) v4i*)(l), 0, 0);
#else
  *(ush8*)l = *(const ush8*)g;
#endif
}

__device__ __forceinline__ void wait_async() {
#if HAVE_ASYNC_LDS
#if __has_builtin(__builtin_amdgcn_s_wait_asynccnt)
  __builtin_amdgcn_s_wait_asynccnt(0);
#else
  asm volatile("s_wait_asynccnt 0x0" ::: "memory");
#endif
#endif
}

__device__ __forceinline__ unsigned short f2bf(float f) {
  unsigned int u = __float_as_uint(f);
  u += 0x7FFFu + ((u >> 16) & 1u);        // round-to-nearest-even
  return (unsigned short)(u >> 16);
}

__device__ __forceinline__ v8f wmma_bf16(const Frag& a, const Frag& b, v8f c) {
  // D = A(16x32 bf16) * B(32x16 bf16) + C(16x16 f32)
  return __builtin_amdgcn_wmma_f32_16x16x32_bf16(
      false, a.v, false, b.v, (short)0, c, false, false);
}

// ---------------- model constants ----------------
constexpr int LAYERS = 12;
constexpr int BATCH  = 4;
constexpr int TSEQ   = 512;
constexpr int NROW   = BATCH * TSEQ;   // 2048
constexpr int CDIM   = 1024;
constexpr int HEADS  = 16;
constexpr int HDIM   = 64;
constexpr int FDIM   = 4096;
constexpr int VOCAB  = 32000;

// ---------------- embedding ----------------
__global__ __launch_bounds__(256) void embed_kernel(
    const int* __restrict__ idx, const float* __restrict__ tok,
    const float* __restrict__ pos, float* __restrict__ x) {
  int gid = blockIdx.x * 256 + threadIdx.x;
  if (gid >= NROW * CDIM) return;
  int n = gid >> 10;          // row
  int c = gid & 1023;
  int t = n & (TSEQ - 1);
  int token = idx[n];
  x[gid] = tok[(size_t)token * CDIM + c] + pos[(size_t)t * CDIM + c];
}

// ---------------- layernorm -> bf16 ----------------
__global__ __launch_bounds__(256) void layernorm_bf16_kernel(
    const float* __restrict__ x, const float* __restrict__ g,
    const float* __restrict__ b, unsigned short* __restrict__ out) {
  int row = blockIdx.x;
  const float* base = x + (size_t)row * CDIM;
  int tid = threadIdx.x;
  __shared__ float r1[256];
  __shared__ float r2[256];
  float s1 = 0.f, s2 = 0.f;
#pragma unroll
  for (int i = 0; i < 4; ++i) {
    float v = base[tid + i * 256];
    s1 += v; s2 += v * v;
  }
  r1[tid] = s1; r2[tid] = s2;
  __syncthreads();
  for (int s = 128; s > 0; s >>= 1) {
    if (tid < s) { r1[tid] += r1[tid + s]; r2[tid] += r2[tid + s]; }
    __syncthreads();
  }
  float mean = r1[0] * (1.f / CDIM);
  float var  = r2[0] * (1.f / CDIM) - mean * mean;
  float rstd = rsqrtf(var + 1e-5f);
#pragma unroll
  for (int i = 0; i < 4; ++i) {
    int c = tid + i * 256;
    float o = (base[c] - mean) * rstd * g[c] + b[c];
    out[(size_t)row * CDIM + c] = f2bf(o);
  }
}

// ---------------- generic WMMA GEMM (double-buffered, async LDS) ----------------
// D(MxN) = A_bf16(MxK) * W_f32->bf16(KxN) [+bias] [relu]; out f32/bf16, opt residual accumulate
template <bool HAS_BIAS, bool RELU, bool RESIDUAL, bool OUT_BF16>
__global__ __launch_bounds__(256) void gemm_wmma_kernel(
    const unsigned short* __restrict__ A, int lda,
    const float* __restrict__ W, int ldw,
    const float* __restrict__ bias,
    float* __restrict__ outF, unsigned short* __restrict__ outB, int ldo,
    int K) {
  const int n0 = blockIdx.x * 128;
  const int m0 = blockIdx.y * 128;
  __shared__ unsigned short As[2][128 * 32];    // [m][k]
  __shared__ unsigned short BsT[2][128 * 32];   // [n][k]  (transposed weight tile)
  const int tid  = threadIdx.x;
  const int lane = tid & 31;
  const int wid  = tid >> 5;
  const int wm   = wid & 3;   // 4 waves along M, 32 rows each
  const int wn   = wid >> 2;  // 2 waves along N, 64 cols each

  v8f acc[2][4];
#pragma unroll
  for (int i = 0; i < 2; ++i)
#pragma unroll
    for (int j = 0; j < 4; ++j)
      acc[i][j] = (v8f){0.f, 0.f, 0.f, 0.f, 0.f, 0.f, 0.f, 0.f};

  const int row   = lane & 15;
  const int khalf = (lane & 16) ? 8 : 0;   // A-operand K sub-block
  const int kb    = (lane & 16) ? 16 : 0;  // unused (B reads also use khalf)

  // stage one K-tile (A: async b128 copies, W: f32 load + bf16 transpose)
  auto stage = [&](int k0, int buf) {
#pragma unroll
    for (int i = 0; i < 2; ++i) {
      int chunk = tid + i * 256;
      int r = chunk >> 2;
      int c = (chunk & 3) << 3;
      async_copy16(&A[(size_t)(m0 + r) * lda + k0 + c], &As[buf][r * 32 + c]);
    }
#pragma unroll
    for (int i = 0; i < 4; ++i) {
      int idx2 = tid + i * 256;
      int kk = idx2 >> 5;
      int nn = (idx2 & 31) << 2;
      const float4 wv4 = *(const float4*)&W[(size_t)(k0 + kk) * ldw + n0 + nn];
      BsT[buf][(nn + 0) * 32 + kk] = f2bf(wv4.x);
      BsT[buf][(nn + 1) * 32 + kk] = f2bf(wv4.y);
      BsT[buf][(nn + 2) * 32 + kk] = f2bf(wv4.z);
      BsT[buf][(nn + 3) * 32 + kk] = f2bf(wv4.w);
    }
  };

  const int KT = K >> 5;
  stage(0, 0);
  wait_async();
  __syncthreads();

  for (int kt = 0; kt < KT; ++kt) {
    const int cur = kt & 1;
    if (kt + 1 < KT) stage((kt + 1) << 5, cur ^ 1);
    if (kt + 2 < KT) {
      // prefetch fp32 weight stream two tiles ahead (global_prefetch_b8)
      int pk = ((kt + 2) << 5) + (tid & 31);
      __builtin_prefetch(&W[(size_t)pk * ldw + n0 + ((tid >> 5) << 4)], 0, 0);
    }

    Frag a[2], b[4];
#pragma unroll
    for (int mt = 0; mt < 2; ++mt) {
      int mr = wm * 32 + mt * 16 + row;
      a[mt].u[0] = *(const ush8*)&As[cur][mr * 32 + khalf];
      a[mt].u[1] = *(const ush8*)&As[cur][mr * 32 + khalf + 16];
    }
#pragma unroll
    for (int nt = 0; nt < 4; ++nt) {
      int nc = wn * 64 + nt * 16 + row;
      b[nt].u[0] = *(const ush8*)&BsT[cur][nc * 32 + khalf];
      b[nt].u[1] = *(const ush8*)&BsT[cur][nc * 32 + khalf + 16];
    }
#pragma unroll
    for (int mt = 0; mt < 2; ++mt)
#pragma unroll
      for (int nt = 0; nt < 4; ++nt)
        acc[mt][nt] = wmma_bf16(a[mt], b[nt], acc[mt][nt]);

    wait_async();
    __syncthreads();
  }

  // epilogue : D layout -> VGPR r holds (M=r, lanes 0-15) / (M=8+r, lanes 16-31)
  const int hi = (lane & 16) ? 8 : 0;
#pragma unroll
  for (int mt = 0; mt < 2; ++mt) {
#pragma unroll
    for (int nt = 0; nt < 4; ++nt) {
      int n = n0 + wn * 64 + nt * 16 + row;
      float bv = HAS_BIAS ? bias[n] : 0.f;
#pragma unroll
      for (int r = 0; r < 8; ++r) {
        int m = m0 + wm * 32 + mt * 16 + r + hi;
        float v = acc[mt][nt][r] + bv;
        if (RELU) v = fmaxf(v, 0.f);
        size_t o = (size_t)m * ldo + n;
        if (RESIDUAL)      outF[o] += v;
        else if (OUT_BF16) outB[o] = f2bf(v);
        else               outF[o] = v;
      }
    }
  }
}

// ---------------- attention: scores = Q K^T (per b,h) ----------------
__global__ __launch_bounds__(256) void attn_scores_kernel(
    const unsigned short* __restrict__ q, const unsigned short* __restrict__ k,
    float* __restrict__ att) {
  const int s0 = blockIdx.x * 64;
  const int t0 = blockIdx.y * 128;
  const int z  = blockIdx.z;          // b*H + h
  if (s0 > t0 + 127) return;          // fully above causal diagonal (uniform)
  const int b = z >> 4, h = z & 15;
  const int lane = threadIdx.x & 31;
  const int wid  = threadIdx.x >> 5;  // 8 waves x 16 rows = 128 rows
  const int row  = lane & 15;
  const int khalf = (lane & 16) ? 8 : 0;
  const int kb    = (lane & 16) ? 16 : 0;
  const int trow = t0 + wid * 16 + row;
  const unsigned short* qrow = q + ((size_t)(b * TSEQ + trow) * CDIM + h * HDIM);

  v8f acc[4];
#pragma unroll
  for (int i = 0; i < 4; ++i) acc[i] = (v8f){0.f,0.f,0.f,0.f,0.f,0.f,0.f,0.f};

#pragma unroll
  for (int kk = 0; kk < HDIM; kk += 32) {
    Frag a;
    a.u[0] = *(const ush8*)&qrow[kk + khalf];
    a.u[1] = *(const ush8*)&qrow[kk + khalf + 16];
#pragma unroll
    for (int nt = 0; nt < 4; ++nt) {
      int s = s0 + nt * 16 + row;
      const unsigned short* kr =
          k + ((size_t)(b * TSEQ + s) * CDIM + h * HDIM + kk + khalf);
      Frag bb;
      bb.u[0] = *(const ush8*)&kr[0];
      bb.u[1] = *(const ush8*)&kr[16];
      acc[nt] = wmma_bf16(a, bb, acc[nt]);
    }
  }
  const int hi = (lane & 16) ? 8 : 0;
#pragma unroll
  for (int nt = 0; nt < 4; ++nt)
#pragma unroll
    for (int r = 0; r < 8; ++r) {
      int t = t0 + wid * 16 + r + hi;
      int s = s0 + nt * 16 + row;
      att[((size_t)z * TSEQ + t) * TSEQ + s] = acc[nt][r];
    }
}

// ---------------- causal softmax (x 1/sqrt(HD)) -> bf16, zero-padded ----------------
__global__ __launch_bounds__(256) void softmax_kernel(
    const float* __restrict__ att, unsigned short* __restrict__ attb) {
  const int rowid = blockIdx.x;         // z*T + t
  const int t = rowid & (TSEQ - 1);
  const float* base = att + (size_t)rowid * TSEQ;
  const int tid = threadIdx.x;
  __shared__ float red[256];

  float lm = -3.4e38f;
  if (tid <= t)        lm = fmaxf(lm, base[tid] * 0.125f);
  if (tid + 256 <= t)  lm = fmaxf(lm, base[tid + 256] * 0.125f);
  red[tid] = lm;
  __syncthreads();
  for (int s = 128; s > 0; s >>= 1) {
    if (tid < s) red[tid] = fmaxf(red[tid], red[tid + s]);
    __syncthreads();
  }
  float m = red[0];
  __syncthreads();

  float p0 = (tid <= t)       ? __expf(base[tid] * 0.125f - m)       : 0.f;
  float p1 = (tid + 256 <= t) ? __expf(base[tid + 256] * 0.125f - m) : 0.f;
  red[tid] = p0 + p1;
  __syncthreads();
  for (int s = 128; s > 0; s >>= 1) {
    if (tid < s) red[tid] += red[tid + s];
    __syncthreads();
  }
  float inv = 1.f / red[0];
  attb[(size_t)rowid * TSEQ + tid]       = f2bf(p0 * inv);
  attb[(size_t)rowid * TSEQ + tid + 256] = f2bf(p1 * inv);
}

// ---------------- attention: Y = P V (per b,h) -> bf16 ----------------
__global__ __launch_bounds__(256) void attn_pv_kernel(
    const unsigned short* __restrict__ attb, const unsigned short* __restrict__ v,
    unsigned short* __restrict__ y) {
  const int t0 = blockIdx.x * 128;
  const int z  = blockIdx.y;
  const int b = z >> 4, h = z & 15;
  const int tid  = threadIdx.x;
  const int lane = tid & 31;
  const int wid  = tid >> 5;
  const int row  = lane & 15;
  const int khalf = (lane & 16) ? 8 : 0;
  const int kb    = (lane & 16) ? 16 : 0;
  const int trow = t0 + wid * 16 + row;
  __shared__ unsigned short VsT[HDIM * 32];   // [hd][s]

  v8f acc[4];
#pragma unroll
  for (int i = 0; i < 4; ++i) acc[i] = (v8f){0.f,0.f,0.f,0.f,0.f,0.f,0.f,0.f};

  const int smax = t0 + 128;                  // causal: probs beyond are exact zeros
  for (int s0 = 0; s0 < smax; s0 += 32) {
    // stage V tile transposed : VsT[hd][k]
    int kk2 = tid >> 3;
    int hdc = (tid & 7) << 3;
    ush8 tmp = *(const ush8*)&v[((size_t)(b * TSEQ + s0 + kk2) * CDIM) + h * HDIM + hdc];
#pragma unroll
    for (int j = 0; j < 8; ++j) VsT[(hdc + j) * 32 + kk2] = tmp[j];
    __syncthreads();

    Frag a;
    const unsigned short* ar = attb + ((size_t)z * TSEQ + trow) * TSEQ + s0;
    a.u[0] = *(const ush8*)&ar[khalf];
    a.u[1] = *(const ush8*)&ar[khalf + 16];
#pragma unroll
    for (int nt = 0; nt < 4; ++nt) {
      int hd = nt * 16 + row;
      Frag bb;
      bb.u[0] = *(const ush8*)&VsT[hd * 32 + khalf];
      bb.u[1] = *(const ush8*)&VsT[hd * 32 + khalf + 16];
      acc[nt] = wmma_bf16(a, bb, acc[nt]);
    }
    __syncthreads();
  }
  const int hi = (lane & 16) ? 8 : 0;
#pragma unroll
  for (int nt = 0; nt < 4; ++nt)
#pragma unroll
    for (int r = 0; r < 8; ++r) {
      int t = t0 + wid * 16 + r + hi;
      int hd = nt * 16 + row;
      y[(size_t)(b * TSEQ + t) * CDIM + h * HDIM + hd] = f2bf(acc[nt][r]);
    }
}

// ---------------- host-side orchestration ----------------
extern "C" void kernel_launch(void* const* d_in, const int* in_sizes, int n_in,
                              void* d_out, int out_size, void* d_ws, size_t ws_size,
                              hipStream_t stream) {
  (void)in_sizes; (void)n_in; (void)out_size; (void)ws_size;
  const int*   idx     = (const int*)d_in[0];
  const float* tok_emb = (const float*)d_in[1];
  const float* pos_emb = (const float*)d_in[2];
  const float* wq = (const float*)d_in[3];
  const float* wk = (const float*)d_in[4];
  const float* wv = (const float*)d_in[5];
  const float* wo = (const float*)d_in[6];
  const float* bo = (const float*)d_in[7];
  const float* ln1_g = (const float*)d_in[8];
  const float* ln1_b = (const float*)d_in[9];
  const float* ln2_g = (const float*)d_in[10];
  const float* ln2_b = (const float*)d_in[11];
  const float* w1 = (const float*)d_in[12];
  const float* b1 = (const float*)d_in[13];
  const float* w2 = (const float*)d_in[14];
  const float* b2 = (const float*)d_in[15];
  const float* lnf_g = (const float*)d_in[16];
  const float* lnf_b = (const float*)d_in[17];
  const float* w_lm  = (const float*)d_in[18];

  // workspace carve-up
  char* p = (char*)d_ws;
  auto carve = [&](size_t bytes) { char* r = p; p += (bytes + 255) & ~(size_t)255; return r; };
  float*          x    = (float*)carve((size_t)NROW * CDIM * 4);
  unsigned short* xn   = (unsigned short*)carve((size_t)NROW * CDIM * 2);
  unsigned short* qb   = (unsigned short*)carve((size_t)NROW * CDIM * 2);
  unsigned short* kb   = (unsigned short*)carve((size_t)NROW * CDIM * 2);
  unsigned short* vb   = (unsigned short*)carve((size_t)NROW * CDIM * 2);
  unsigned short* yb   = (unsigned short*)carve((size_t)NROW * CDIM * 2);
  unsigned short* hb   = (unsigned short*)carve((size_t)NROW * FDIM * 2);
  float*          att  = (float*)carve((size_t)BATCH * HEADS * TSEQ * TSEQ * 4);
  unsigned short* attb = (unsigned short*)carve((size_t)BATCH * HEADS * TSEQ * TSEQ * 2);

  const dim3 blk(256);
  const dim3 gEmb((NROW * CDIM + 255) / 256);
  const dim3 gLN(NROW);
  const dim3 gC(CDIM / 128, NROW / 128);     // N=1024 GEMMs
  const dim3 gF(FDIM / 128, NROW / 128);     // N=4096 GEMM
  const dim3 gV(VOCAB / 128, NROW / 128);    // N=32000 GEMM
  const dim3 gSc(TSEQ / 64, TSEQ / 128, BATCH * HEADS);
  const dim3 gSm(BATCH * HEADS * TSEQ);
  const dim3 gPV(TSEQ / 128, BATCH * HEADS);

  embed_kernel<<<gEmb, blk, 0, stream>>>(idx, tok_emb, pos_emb, x);

  for (int l = 0; l < LAYERS; ++l) {
    const size_t wOff = (size_t)l * CDIM * CDIM;
    layernorm_bf16_kernel<<<gLN, blk, 0, stream>>>(x, ln1_g + l * CDIM, ln1_b + l * CDIM, xn);
    gemm_wmma_kernel<false, false, false, true><<<gC, blk, 0, stream>>>(
        xn, CDIM, wq + wOff, CDIM, nullptr, nullptr, qb, CDIM, CDIM);
    gemm_wmma_kernel<false, false, false, true><<<gC, blk, 0, stream>>>(
        xn, CDIM, wk + wOff, CDIM, nullptr, nullptr, kb, CDIM, CDIM);
    gemm_wmma_kernel<false, false, false, true><<<gC, blk, 0, stream>>>(
        xn, CDIM, wv + wOff, CDIM, nullptr, nullptr, vb, CDIM, CDIM);

    attn_scores_kernel<<<gSc, blk, 0, stream>>>(qb, kb, att);
    softmax_kernel<<<gSm, blk, 0, stream>>>(att, attb);
    attn_pv_kernel<<<gPV, blk, 0, stream>>>(attb, vb, yb);

    gemm_wmma_kernel<true, false, true, false><<<gC, blk, 0, stream>>>(
        yb, CDIM, wo + wOff, CDIM, bo + l * CDIM, x, nullptr, CDIM, CDIM);

    layernorm_bf16_kernel<<<gLN, blk, 0, stream>>>(x, ln2_g + l * CDIM, ln2_b + l * CDIM, xn);
    gemm_wmma_kernel<true, true, false, true><<<gF, blk, 0, stream>>>(
        xn, CDIM, w1 + (size_t)l * CDIM * FDIM, FDIM, b1 + (size_t)l * FDIM,
        nullptr, hb, FDIM, CDIM);
    gemm_wmma_kernel<true, false, true, false><<<gC, blk, 0, stream>>>(
        hb, FDIM, w2 + (size_t)l * FDIM * CDIM, CDIM, b2 + l * CDIM,
        x, nullptr, CDIM, FDIM);
  }

  layernorm_bf16_kernel<<<gLN, blk, 0, stream>>>(x, lnf_g, lnf_b, xn);
  gemm_wmma_kernel<false, false, false, false><<<gV, blk, 0, stream>>>(
      xn, CDIM, w_lm, VOCAB, nullptr, (float*)d_out, nullptr, VOCAB, CDIM);
}
